// PCTSegment_49804440765001
// MI455X (gfx1250) — compile-verified, matrix operand order checked by reading, not run on hardware
//
#include <hip/hip_runtime.h>

// ---------------------------------------------------------------- types/dims
typedef _Float16 v16h __attribute__((ext_vector_type(16)));
typedef _Float16 h8   __attribute__((ext_vector_type(8)));
typedef float    v8f  __attribute__((ext_vector_type(8)));
typedef unsigned int u32x4 __attribute__((ext_vector_type(4)));
typedef int          i32x4 __attribute__((ext_vector_type(4)));
typedef int          i32x8 __attribute__((ext_vector_type(8)));

constexpr int kB  = 16;
constexpr int kN  = 2048;
constexpr int kC  = 512;
constexpr int kC4 = 128;
constexpr float kEPS = 1e-5f;

#define TM 128
#define TN 128
#define TK 32

// Tensor Data Mover path for the A tile (gfx1250). Guarded so the file still
// compiles on toolchains lacking the builtin; arity differs between clang-22
// (ROCm 7.2: 5 args) and clang-23 (therock: 6 args).
#if defined(__gfx1250__) && __has_builtin(__builtin_amdgcn_tensor_load_to_lds) && \
    __has_builtin(__builtin_amdgcn_s_wait_tensorcnt)
#define USE_TDM 1
#else
#define USE_TDM 0
#endif

// ---------------------------------------------------------------- f32 -> f16
__global__ __launch_bounds__(256) void k_cvt_f16(_Float16* __restrict__ d,
                                                 const float* __restrict__ s) {
  size_t i = (size_t)blockIdx.x * 256 + threadIdx.x;
  d[i] = (_Float16)s[i];
}

// ------------------------------------------------------- encoder first conv
__global__ __launch_bounds__(256) void k_enc1(const float* __restrict__ x,
                                              const float* __restrict__ w,
                                              float* __restrict__ out) {
  size_t idx = (size_t)blockIdx.x * 256 + threadIdx.x;
  int b = idx / ((size_t)kC * kN);
  size_t r = idx % ((size_t)kC * kN);
  int d = r / kN, n = r % kN;
  const float* xp = x + ((size_t)b * kN + n) * 3;
  const float* wp = w + (size_t)d * 3;
  out[idx] = wp[0] * xp[0] + wp[1] * xp[1] + wp[2] * xp[2];
}

// ------------------------------------------------- generic batch-norm stats
__global__ __launch_bounds__(256) void k_bn_stats(const float* __restrict__ src,
                                                  float* __restrict__ mean,
                                                  float* __restrict__ rstd,
                                                  int J, long bs, long js, long cs) {
  int c = blockIdx.x, tid = threadIdx.x;
  int total = kB * J;
  float s = 0.f, s2 = 0.f;
  for (int idx = tid; idx < total; idx += 256) {
    int b = idx / J, j = idx % J;
    float v = src[(size_t)b * bs + (size_t)j * js + (size_t)c * cs];
    s += v; s2 += v * v;
  }
  __shared__ float sh[256], sh2[256];
  sh[tid] = s; sh2[tid] = s2; __syncthreads();
  for (int o = 128; o > 0; o >>= 1) {
    if (tid < o) { sh[tid] += sh[tid + o]; sh2[tid] += sh2[tid + o]; }
    __syncthreads();
  }
  if (tid == 0) {
    float m = sh[0] / total;
    float var = sh2[0] / total - m * m;
    mean[c] = m;
    rstd[c] = rsqrtf(var + kEPS);
  }
}

// ------------- fused BN apply + activation + residual + multi-dest scatter
__global__ __launch_bounds__(256) void k_bn_apply(
    const float* __restrict__ src, const float* __restrict__ mean,
    const float* __restrict__ rstd, const float* __restrict__ gam,
    const float* __restrict__ bet, const float* __restrict__ res, int act,
    int CHn, int L,
    float* o32a, long bsA, float* o32b, long bsB,
    _Float16* o16a, long bsC, _Float16* o16b, long bsD) {
  size_t idx = (size_t)blockIdx.x * 256 + threadIdx.x;
  size_t per = (size_t)CHn * L;
  int b = idx / per;
  size_t r = idx % per;
  int ch = r / L;
  float v = src[idx];
  v = gam[ch] * (v - mean[ch]) * rstd[ch] + bet[ch];
  if (act == 1) v = fmaxf(v, 0.f);
  else if (act == 2) v = v > 0.f ? v : 0.2f * v;
  if (res) v += res[idx];
  if (o32a) o32a[(size_t)b * bsA + r] = v;
  if (o32b) o32b[(size_t)b * bsB + r] = v;
  if (o16a) o16a[(size_t)b * bsC + r] = (_Float16)v;
  if (o16b) o16b[(size_t)b * bsD + r] = (_Float16)v;
}

// -------- per-point BN (lbr layers): src [B, 256, N], channel = n (last dim)
__global__ __launch_bounds__(256) void k_bn_apply_pern(
    const float* __restrict__ src, const float* __restrict__ mean,
    const float* __restrict__ rstd, const float* __restrict__ gam,
    const float* __restrict__ bet,
    float* __restrict__ o32, _Float16* __restrict__ o16) {
  size_t idx = (size_t)blockIdx.x * 256 + threadIdx.x;
  int n = idx & (kN - 1);
  float v = src[idx];
  v = gam[n] * (v - mean[n]) * rstd[n] + bet[n];
  v = v > 0.f ? v : 0.2f * v;  // leaky relu
  if (o32) o32[idx] = v;
  if (o16) o16[idx] = (_Float16)v;
}

// --------------------------------------------------------- WMMA f16 GEMM
// C[b] = A[b] x B[b] (+bias per row). A:[M,K] (transA: element = A[k*lda+m]),
// B:[K,Nn] ldb, C f32 ldc (+ optional f16 mirror). All dims multiples of the
// tile -> no predication, EXEC stays all-ones (WMMA requirement).
// A tile is DMA'd by the Tensor Data Mover (wave 0 issues; TDM pad_enable
// reproduces the 32+8-half padded LDS rows); B tile is staged transposed by
// the vector pipe (TDM cannot transpose).
__global__ __launch_bounds__(256) void k_gemm_wmma(
    const _Float16* __restrict__ Ag, const _Float16* __restrict__ Bg,
    float* __restrict__ Cg, _Float16* __restrict__ C16,
    const float* __restrict__ bias,
    int M, int Nn, int K, int lda, int ldb, int ldc,
    long bsA, long bsB, long bsC, long bsC16, int transA) {
  __shared__ _Float16 lA[TM][TK + 8];   // row stride 40 halves = 80B (16B aligned)
  __shared__ _Float16 lBt[TN][TK + 8];  // B tile stored transposed [n][k]

  int bz = blockIdx.z;
  const _Float16* A = Ag + (size_t)bz * bsA;
  const _Float16* Bp = Bg + (size_t)bz * bsB;
  float* Cp = Cg + (size_t)bz * bsC;
  _Float16* Hp = C16 ? C16 + (size_t)bz * bsC16 : (_Float16*)0;

  int m0 = blockIdx.y * TM, n0 = blockIdx.x * TN;
  int tid = threadIdx.x;
  int lane = tid & 31, wave = tid >> 5;
  int wrow = wave & 3, wcol = wave >> 2;   // 4x2 waves, each owns 32x64
  int hsel = lane >> 4, rr16 = lane & 15;

  int ar = tid >> 1, seg = tid & 1;        // A staging coords
  int kk = tid & 31, nseg = tid >> 5;      // B staging coords

  const _Float16* pa;
  long stepA;
  if (!transA) { pa = A + (size_t)(m0 + ar) * lda + seg * 16; stepA = TK; }
  else         { pa = A + (size_t)(seg * 16) * lda + (m0 + ar); stepA = (long)TK * lda; }
  const _Float16* pb = Bp + (size_t)kk * ldb + n0 + nseg * 16;
  const long stepB = (long)TK * ldb;

#if USE_TDM
  unsigned long long gA = (unsigned long long)(uintptr_t)(A + (size_t)m0 * lda);
  unsigned ldsA = (unsigned)(uintptr_t)&lA[0][0];  // low 32 bits = LDS offset
#endif

  v8f acc[2][4];
#pragma unroll
  for (int i = 0; i < 2; ++i)
#pragma unroll
    for (int j = 0; j < 4; ++j) acc[i][j] = (v8f){0, 0, 0, 0, 0, 0, 0, 0};

  for (int k0 = 0; k0 < K; k0 += TK) {
    const bool last = (k0 + TK >= K);
    // ---- stage A tile: 128x32 halves
#if USE_TDM
    if (!transA) {
      if (tid < 32) {  // one wave issues the DMA; TDM ignores EXEC
        unsigned long long ga = gA + (unsigned long long)k0 * 2;
        u32x4 g0;
        g0[0] = 1u;                                  // count = 1 (valid D#)
        g0[1] = ldsA;                                // LDS destination
        g0[2] = (unsigned)ga;                        // global_addr[31:0]
        g0[3] = (unsigned)((ga >> 32) & 0x1ffffffu)  // global_addr[56:32]
                | (2u << 30);                        // type = 2 ("image")
        i32x8 g1;
        g1[0] = (int)((1u << 16)      // data_size = 2B
                      | (1u << 20)    // pad_enable (LDS row padding)
                      | (3u << 22)    // pad_interval: every 16 DWORDs (one row)
                      | (3u << 25));  // pad_amount: 4 DWORDs (8 halves)
        g1[1] = (int)(((unsigned)K & 0xffffu) << 16);                    // dim0 lo
        g1[2] = (int)(((unsigned)K >> 16) | (((unsigned)M & 0xffffu) << 16));
        g1[3] = (int)(((unsigned)M >> 16) | ((unsigned)TK << 16));       // tile_dim0
        g1[4] = TM;                                  // tile_dim1 (tile_dim2 = 0)
        g1[5] = lda;                                 // tensor_dim0_stride lo32
        g1[6] = 0; g1[7] = 0;
        i32x4 g2 = (i32x4){0, 0, 0, 0}, g3 = (i32x4){0, 0, 0, 0};
#if __clang_major__ >= 23
        __builtin_amdgcn_tensor_load_to_lds(g0, g1, g2, g3,
                                            (i32x8){0, 0, 0, 0, 0, 0, 0, 0}, 0);
#else
        __builtin_amdgcn_tensor_load_to_lds(g0, g1, g2, g3, 0);
#endif
      }
    } else {
      const _Float16* ap = pa;   // additive addressing (no per-element 64b mul)
#pragma unroll
      for (int jj = 0; jj < 16; ++jj) { lA[ar][seg * 16 + jj] = *ap; ap += lda; }
    }
#else
    if (!transA) {
      const h8* p = (const h8*)pa;
      h8 x0 = p[0], x1 = p[1];
      *(h8*)&lA[ar][seg * 16] = x0;
      *(h8*)&lA[ar][seg * 16 + 8] = x1;
      if (!last) __builtin_prefetch(pa + TK, 0, 1);
    } else {
      const _Float16* ap = pa;
#pragma unroll
      for (int jj = 0; jj < 16; ++jj) { lA[ar][seg * 16 + jj] = *ap; ap += lda; }
    }
#endif
    pa += stepA;

    // ---- stage B tile transposed: coalesced global reads, LDS [n][k]
    {
      const h8* p = (const h8*)pb;
      h8 x0 = p[0], x1 = p[1];
#pragma unroll
      for (int jj = 0; jj < 8; ++jj) {
        lBt[nseg * 16 + jj][kk] = x0[jj];
        lBt[nseg * 16 + 8 + jj][kk] = x1[jj];
      }
      if (!last) __builtin_prefetch(pb + stepB, 0, 1);
      pb += stepB;
    }

#if USE_TDM
    if (!transA && tid < 32) __builtin_amdgcn_s_wait_tensorcnt(0);
#endif
    __syncthreads();

    // ---- fragments: 16-bit A 16x32 layout => lane group hsel holds
    // K in {hsel*8 .. hsel*8+7} U {hsel*8+16 .. hsel*8+23}: two 16B LDS reads
    v16h af[2], bf[4];
#pragma unroll
    for (int ms = 0; ms < 2; ++ms) {
      const _Float16* fp = &lA[wrow * 32 + ms * 16 + rr16][hsel * 8];
      h8 lo = *(const h8*)fp;
      h8 hi = *(const h8*)(fp + 16);
      af[ms] = __builtin_shufflevector(lo, hi, 0, 1, 2, 3, 4, 5, 6, 7,
                                       8, 9, 10, 11, 12, 13, 14, 15);
    }
#pragma unroll
    for (int ns = 0; ns < 4; ++ns) {
      const _Float16* fp = &lBt[wcol * 64 + ns * 16 + rr16][hsel * 8];
      h8 lo = *(const h8*)fp;
      h8 hi = *(const h8*)(fp + 16);
      bf[ns] = __builtin_shufflevector(lo, hi, 0, 1, 2, 3, 4, 5, 6, 7,
                                       8, 9, 10, 11, 12, 13, 14, 15);
    }
#pragma unroll
    for (int ms = 0; ms < 2; ++ms)
#pragma unroll
      for (int ns = 0; ns < 4; ++ns)
        acc[ms][ns] = __builtin_amdgcn_wmma_f32_16x16x32_f16(
            false, af[ms], false, bf[ns], (short)0, acc[ms][ns], false, false);
    __syncthreads();
  }

  // ---- epilogue
#pragma unroll
  for (int ms = 0; ms < 2; ++ms)
#pragma unroll
    for (int ns = 0; ns < 4; ++ns) {
      int row0 = m0 + wrow * 32 + ms * 16 + hsel * 8;
      int col = n0 + wcol * 64 + ns * 16 + rr16;
#pragma unroll
      for (int i = 0; i < 8; ++i) {
        int row = row0 + i;
        float v = acc[ms][ns][i];
        if (bias) v += bias[row];
        Cp[(size_t)row * ldc + col] = v;
        if (Hp) Hp[(size_t)row * ldc + col] = (_Float16)v;
      }
    }
}

// -------------------------------------------- masked softmax over last axis
__global__ __launch_bounds__(256) void k_softmax(float* __restrict__ att,
                                                 const int* __restrict__ mask) {
  int bn = blockIdx.x;
  int b = bn >> 11, n = bn & (kN - 1);
  float* row = att + (size_t)bn * kN;
  int mrow = mask[(b << 11) | n];
  const int* mcol = mask + ((size_t)b << 11);
  int tid = threadIdx.x;
  float e[8], mx = -3.4e38f;
#pragma unroll
  for (int k = 0; k < 8; ++k) {
    int m = tid + (k << 8);
    float v = (mrow && mcol[m]) ? row[m] : -9e15f;
    e[k] = v;
    mx = fmaxf(mx, v);
  }
  __shared__ float sh[256];
  sh[tid] = mx; __syncthreads();
  for (int o = 128; o > 0; o >>= 1) {
    if (tid < o) sh[tid] = fmaxf(sh[tid], sh[tid + o]);
    __syncthreads();
  }
  mx = sh[0];
  __syncthreads();
  float s = 0.f;
#pragma unroll
  for (int k = 0; k < 8; ++k) { e[k] = expf(e[k] - mx); s += e[k]; }
  sh[tid] = s; __syncthreads();
  for (int o = 128; o > 0; o >>= 1) {
    if (tid < o) sh[tid] += sh[tid + o];
    __syncthreads();
  }
  float inv = 1.0f / sh[0];
#pragma unroll
  for (int k = 0; k < 8; ++k) row[tid + (k << 8)] = e[k] * inv;
}

// column sums over dim n: colsum[b,m] = sum_n att[b,n,m]
__global__ __launch_bounds__(256) void k_colsum(const float* __restrict__ att,
                                                float* __restrict__ colsum) {
  int bm = blockIdx.x * 256 + threadIdx.x;
  int b = bm >> 11, m = bm & (kN - 1);
  const float* base = att + (size_t)b * kN * kN + m;
  float s = 0.f;
  for (int n = 0; n < kN; ++n) s += base[(size_t)n * kN];
  colsum[bm] = s;
}

// att16 = att / (1e-9 + colsum[b,m])
__global__ __launch_bounds__(256) void k_attscale(const float* __restrict__ att,
                                                  const float* __restrict__ colsum,
                                                  _Float16* __restrict__ att16) {
  size_t idx = (size_t)blockIdx.x * 256 + threadIdx.x;
  int b = idx >> 22;  // N*N = 2^22
  int m = idx & (kN - 1);
  att16[idx] = (_Float16)(att[idx] / (1e-9f + colsum[(b << 11) | m]));
}

// xm16 = (f16)(h - xr)
__global__ __launch_bounds__(256) void k_sub_f16(const float* __restrict__ h,
                                                 const float* __restrict__ xr,
                                                 _Float16* __restrict__ xm) {
  size_t idx = (size_t)blockIdx.x * 256 + threadIdx.x;
  xm[idx] = (_Float16)(h[idx] - xr[idx]);
}

// row max over n: one block per (b, j)
__global__ __launch_bounds__(256) void k_rowmax(const float* __restrict__ xcat,
                                                float* __restrict__ y0) {
  const float* row = xcat + (size_t)blockIdx.x * kN;
  int tid = threadIdx.x;
  float mx = -3.4e38f;
#pragma unroll
  for (int k = 0; k < 8; ++k) mx = fmaxf(mx, row[tid + (k << 8)]);
  __shared__ float sh[256];
  sh[tid] = mx; __syncthreads();
  for (int o = 128; o > 0; o >>= 1) {
    if (tid < o) sh[tid] = fmaxf(sh[tid], sh[tid + o]);
    __syncthreads();
  }
  if (tid == 0) y0[blockIdx.x] = sh[0];
}

// y1[b,i] = dot(dg_w[i,:], y0[b,:])
__global__ __launch_bounds__(256) void k_dg(const float* __restrict__ y0,
                                            const float* __restrict__ dgw,
                                            float* __restrict__ y1) {
  int t = blockIdx.x * 256 + threadIdx.x;
  int b = t >> 11, i = t & (kN - 1);
  const float* w = dgw + (size_t)i * kN;
  const float* v = y0 + ((size_t)b << 11);
  float s = 0.f;
  for (int j = 0; j < kN; ++j) s += w[j] * v[j];
  y1[t] = s;
}

// per-feature BN over batch axis only (16 samples), lrelu, f16
__global__ __launch_bounds__(256) void k_dg_bn(const float* __restrict__ y1,
                                               const float* __restrict__ gam,
                                               const float* __restrict__ bet,
                                               _Float16* __restrict__ y16) {
  int i = blockIdx.x * 256 + threadIdx.x;
  float s = 0.f, s2 = 0.f;
  for (int b = 0; b < kB; ++b) {
    float v = y1[(b << 11) | i];
    s += v; s2 += v * v;
  }
  float m = s / kB;
  float rs = rsqrtf(s2 / kB - m * m + kEPS);
  for (int b = 0; b < kB; ++b) {
    float v = gam[i] * (y1[(b << 11) | i] - m) * rs + bet[i];
    v = v > 0.f ? v : 0.2f * v;
    y16[(b << 11) | i] = (_Float16)v;
  }
}

// broadcast y16[b,i] into in16[b, 2048+i, n]
__global__ __launch_bounds__(256) void k_bcast(const _Float16* __restrict__ y16,
                                               _Float16* __restrict__ in16) {
  size_t idx = (size_t)blockIdx.x * 256 + threadIdx.x;
  int b = idx / ((size_t)kN * kN);
  size_t r = idx % ((size_t)kN * kN);
  int i = r / kN, n = r & (kN - 1);
  in16[((size_t)b * 4096 + 2048 + i) * kN + n] = y16[(b << 11) | i];
}

// out[b,n] = sum_o ow[o]*h[b,o,n] + ob
__global__ __launch_bounds__(256) void k_final(const float* __restrict__ h,
                                               const float* __restrict__ ow,
                                               const float* __restrict__ ob,
                                               float* __restrict__ out) {
  int t = blockIdx.x * 256 + threadIdx.x;
  int b = t >> 11, n = t & (kN - 1);
  const float* base = h + (size_t)b * 256 * kN + n;
  float s = 0.f;
  for (int o = 0; o < 256; ++o) s += ow[o] * base[(size_t)o * kN];
  out[t] = s + ob[0];
}

// =============================================================== launcher
extern "C" void kernel_launch(void* const* d_in, const int* in_sizes, int n_in,
                              void* d_out, int out_size, void* d_ws, size_t ws_size,
                              hipStream_t stream) {
  const float* x       = (const float*)d_in[0];
  const int*   mask    = (const int*)d_in[1];
  const float* enc_w1  = (const float*)d_in[2];
  const float* enc_g1  = (const float*)d_in[3];
  const float* enc_b1  = (const float*)d_in[4];
  const float* enc_w2  = (const float*)d_in[5];
  const float* enc_g2  = (const float*)d_in[6];
  const float* enc_b2  = (const float*)d_in[7];
  const float* pt_w1   = (const float*)d_in[8];
  const float* pt_g1   = (const float*)d_in[9];
  const float* pt_b1   = (const float*)d_in[10];
  const float* pt_w2   = (const float*)d_in[11];
  const float* pt_g2   = (const float*)d_in[12];
  const float* pt_b2   = (const float*)d_in[13];
  const float* sa_wqk  = (const float*)d_in[14];
  const float* sa_wv   = (const float*)d_in[15];
  const float* sa_bv   = (const float*)d_in[16];
  const float* sa_wt   = (const float*)d_in[17];
  const float* sa_bt   = (const float*)d_in[18];
  const float* sa_g    = (const float*)d_in[19];
  const float* sa_b    = (const float*)d_in[20];
  const float* dg_w    = (const float*)d_in[21];
  const float* dg_g    = (const float*)d_in[22];
  const float* dg_b    = (const float*)d_in[23];
  const float* l1_w    = (const float*)d_in[24];
  const float* l1_b    = (const float*)d_in[25];
  const float* l1_g    = (const float*)d_in[26];
  const float* l1_bb   = (const float*)d_in[27];
  const float* l2_w    = (const float*)d_in[28];
  const float* l2_b    = (const float*)d_in[29];
  const float* l2_g    = (const float*)d_in[30];
  const float* l2_bb   = (const float*)d_in[31];
  const float* out_w   = (const float*)d_in[32];
  const float* out_b   = (const float*)d_in[33];
  float* outp = (float*)d_out;

  // ---- workspace carve
  size_t off = 0;
  char* base = (char*)d_ws;
  auto carve = [&](size_t bytes) -> char* {
    char* p = base + off;
    off += (bytes + 255) & ~(size_t)255;
    return p;
  };
  const size_t eBCN = (size_t)kB * kC * kN;
  const size_t eBNN = (size_t)kB * kN * kN;
  float*    H32   = (float*)carve(eBCN * 4);
  float*    T32   = (float*)carve(eBCN * 4);
  _Float16* H16   = (_Float16*)carve(eBCN * 2);
  _Float16* XM16  = (_Float16*)carve(eBCN * 2);
  _Float16* V16   = (_Float16*)carve(eBCN * 2);
  _Float16* Q16   = (_Float16*)carve((size_t)kB * kC4 * kN * 2);
  float*    ATT32 = (float*)carve(eBNN * 4);
  _Float16* ATT16 = (_Float16*)carve(eBNN * 2);
  float*    XCAT  = (float*)carve(eBNN * 4);            // [B,4C,N], 4C==N
  _Float16* IN16  = (_Float16*)carve((size_t)kB * 4096 * kN * 2);
  float*    MEANC = (float*)carve(kC * 4);
  float*    RSTDC = (float*)carve(kC * 4);
  float*    MEANN = (float*)carve(kN * 4);
  float*    RSTDN = (float*)carve(kN * 4);
  float*    CSUM  = (float*)carve((size_t)kB * kN * 4);
  float*    Y0    = (float*)carve((size_t)kB * kN * 4);
  float*    Y1    = (float*)carve((size_t)kB * kN * 4);
  _Float16* Y16   = (_Float16*)carve((size_t)kB * kN * 2);
  float*    L1    = (float*)carve((size_t)kB * 256 * kN * 4);
  _Float16* L1H   = (_Float16*)carve((size_t)kB * 256 * kN * 2);
  float*    L2    = (float*)carve((size_t)kB * 256 * kN * 4);
  float*    L2N   = (float*)carve((size_t)kB * 256 * kN * 4);
  _Float16* W2H   = (_Float16*)carve((size_t)kC * kC * 2);
  _Float16* PT1H  = (_Float16*)carve((size_t)kC * kC * 2);
  _Float16* PT2H  = (_Float16*)carve((size_t)kC * kC * 2);
  _Float16* WQKH  = (_Float16*)carve((size_t)4 * kC4 * kC * 2);
  _Float16* WVH   = (_Float16*)carve((size_t)4 * kC * kC * 2);
  _Float16* WTH   = (_Float16*)carve((size_t)4 * kC * kC * 2);
  _Float16* L1WH  = (_Float16*)carve((size_t)256 * 4096 * 2);
  _Float16* L2WH  = (_Float16*)carve((size_t)256 * 256 * 2);

  auto cvt = [&](_Float16* d, const float* s, size_t n) {
    k_cvt_f16<<<(unsigned)(n / 256), 256, 0, stream>>>(d, s);
  };
  cvt(W2H, enc_w2, (size_t)kC * kC);
  cvt(PT1H, pt_w1, (size_t)kC * kC);
  cvt(PT2H, pt_w2, (size_t)kC * kC);
  cvt(WQKH, sa_wqk, (size_t)4 * kC4 * kC);
  cvt(WVH, sa_wv, (size_t)4 * kC * kC);
  cvt(WTH, sa_wt, (size_t)4 * kC * kC);
  cvt(L1WH, l1_w, (size_t)256 * 4096);
  cvt(L2WH, l2_w, (size_t)256 * 256);

  auto gemm = [&](const _Float16* A, const _Float16* Bm, float* Cm, _Float16* C16m,
                  const float* bias, int M, int Nn, int K, int lda, int ldb, int ldc,
                  long bsA, long bsB, long bsC, long bsC16, int transA) {
    dim3 g(Nn / TN, M / TM, kB);
    k_gemm_wmma<<<g, 256, 0, stream>>>(A, Bm, Cm, C16m, bias, M, Nn, K, lda, ldb,
                                       ldc, bsA, bsB, bsC, bsC16, transA);
  };
  auto statsC = [&](const float* s) {
    k_bn_stats<<<kC, 256, 0, stream>>>(s, MEANC, RSTDC, kN, (long)kC * kN, 1L, (long)kN);
  };
  auto statsPN = [&](const float* s) {
    k_bn_stats<<<kN, 256, 0, stream>>>(s, MEANN, RSTDN, 256, (long)256 * kN, (long)kN, 1L);
  };
  const unsigned gBCN = (unsigned)(eBCN / 256);
  auto applyC = [&](const float* g, const float* bb, const float* res, int act,
                    float* oA, long bsA2, float* oB, long bsB2,
                    _Float16* oC, long bsC2, _Float16* oD, long bsD2) {
    k_bn_apply<<<gBCN, 256, 0, stream>>>(T32, MEANC, RSTDC, g, bb, res, act,
                                         kC, kN, oA, bsA2, oB, bsB2, oC, bsC2, oD, bsD2);
  };

  const long sCN = (long)kC * kN;
  const long sNN = (long)kN * kN;

  // ---------------- encoder: Conv(3->C) BN LReLU ; Conv(C->C) BN LReLU
  k_enc1<<<gBCN, 256, 0, stream>>>(x, enc_w1, T32);
  statsC(T32);
  applyC(enc_g1, enc_b1, nullptr, 2, H32, sCN, nullptr, 0, H16, sCN, nullptr, 0);

  gemm(W2H, H16, T32, nullptr, nullptr, kC, kN, kC, kC, kN, kN, 0, sCN, sCN, 0, 0);
  statsC(T32);
  applyC(enc_g2, enc_b2, nullptr, 2, H32, sCN, nullptr, 0, H16, sCN, nullptr, 0);

  // ---------------- pt_last pre-convs (ReLU)
  gemm(PT1H, H16, T32, nullptr, nullptr, kC, kN, kC, kC, kN, kN, 0, sCN, sCN, 0, 0);
  statsC(T32);
  applyC(pt_g1, pt_b1, nullptr, 1, H32, sCN, nullptr, 0, H16, sCN, nullptr, 0);

  gemm(PT2H, H16, T32, nullptr, nullptr, kC, kN, kC, kC, kN, kN, 0, sCN, sCN, 0, 0);
  statsC(T32);
  applyC(pt_g2, pt_b2, nullptr, 1, H32, sCN, nullptr, 0, H16, sCN, nullptr, 0);

  // ---------------- 4 offset self-attention layers
  for (int i = 0; i < 4; ++i) {
    gemm(WQKH + (size_t)i * kC4 * kC, H16, T32, Q16, nullptr,
         kC4, kN, kC, kC, kN, kN, 0, sCN, sCN, (long)kC4 * kN, 0);
    gemm(Q16, Q16, ATT32, nullptr, nullptr,
         kN, kN, kC4, kN, kN, kN, (long)kC4 * kN, (long)kC4 * kN, sNN, 0, 1);
    k_softmax<<<kB * kN, 256, 0, stream>>>(ATT32, mask);
    k_colsum<<<(kB * kN) / 256, 256, 0, stream>>>(ATT32, CSUM);
    k_attscale<<<(unsigned)(eBNN / 256), 256, 0, stream>>>(ATT32, CSUM, ATT16);
    gemm(WVH + (size_t)i * kC * kC, H16, T32, V16, sa_bv + (size_t)i * kC,
         kC, kN, kC, kC, kN, kN, 0, sCN, sCN, sCN, 0);
    gemm(V16, ATT16, T32, nullptr, nullptr,
         kC, kN, kN, kN, kN, kN, sCN, sNN, sCN, 0, 0);
    k_sub_f16<<<gBCN, 256, 0, stream>>>(H32, T32, XM16);
    gemm(WTH + (size_t)i * kC * kC, XM16, T32, nullptr, sa_bt + (size_t)i * kC,
         kC, kN, kC, kC, kN, kN, 0, sCN, sCN, 0, 0);
    statsC(T32);
    applyC(sa_g + (size_t)i * kC, sa_b + (size_t)i * kC, H32, 1,
           H32, sCN, XCAT + (size_t)i * kC * kN, (long)4 * kC * kN,
           H16, sCN, IN16 + (size_t)i * kC * kN, (long)4096 * kN);
  }

  // ---------------- global branch
  k_rowmax<<<kB * kN, 256, 0, stream>>>(XCAT, Y0);
  k_dg<<<(kB * kN) / 256, 256, 0, stream>>>(Y0, dg_w, Y1);
  k_dg_bn<<<kN / 256, 256, 0, stream>>>(Y1, dg_g, dg_b, Y16);
  k_bcast<<<(unsigned)(eBNN / 256), 256, 0, stream>>>(Y16, IN16);

  // ---------------- lbr1
  gemm(L1WH, IN16, L1, nullptr, l1_b, 256, kN, 4096, 4096, kN, kN,
       0, (long)4096 * kN, (long)256 * kN, 0, 0);
  statsPN(L1);
  k_bn_apply_pern<<<(unsigned)((size_t)kB * 256 * kN / 256), 256, 0, stream>>>(
      L1, MEANN, RSTDN, l1_g, l1_bb, nullptr, L1H);

  // ---------------- lbr2
  gemm(L2WH, L1H, L2, nullptr, l2_b, 256, kN, 256, 256, kN, kN,
       0, (long)256 * kN, (long)256 * kN, 0, 0);
  statsPN(L2);
  k_bn_apply_pern<<<(unsigned)((size_t)kB * 256 * kN / 256), 256, 0, stream>>>(
      L2, MEANN, RSTDN, l2_g, l2_bb, L2N, nullptr);

  // ---------------- linear_last
  k_final<<<(kB * kN) / 256, 256, 0, stream>>>(L2N, out_w, out_b, outp);
}
